// aTENNuate_38053410242663
// MI455X (gfx1250) — matrix operand order repaired and checked by reading, not compiled, not run on hardware
//
#include <hip/hip_runtime.h>

// =====================================================================
// aTENNuate SSM U-Net forward for MI455X (gfx1250, wave32).
//
// Strategy (bandwidth-roofline driven, 23.3 TB/s HBM):
//   * SSM long conv == first-order complex linear recurrence
//         h_t = lam * h_{t-1} + (B_hat @ x)_t ,  y = C @ Re(h)
//     implemented as a 3-phase chunked scan (chunk=64) so the (B,N,T)
//     complex state (>1 GB at T=65536) is never materialized in HBM.
//   * The C-projection (c_out x 256) @ (256 x 64) per chunk is done with
//     v_wmma_f32_16x16x32_bf16. C is pre-converted to bf16 once per layer
//     (reused by ~8K chunk-tiles); Re(h) is staged transposed in LDS as
//     bf16 (33 KB) so both fragments load as contiguous b128s.
//   * Everything else (dwconv3, LN+SiLU, mix einsums) is cheap VALU.
//
// Input flattening assumption: setup_inputs() dict insertion order:
//   x, down[0..5]{block{(conv_w,conv_b)?,ssm{A,B,log_dt,C},(ln_w,ln_b)?},mix},
//   up[0..5]{mix,block{...}}, hid[0..1]{block}, last[0..1]{ssm only}.
// Workspace use ~83 MB.
// =====================================================================

#define CHUNK   64
#define NSTATE  256
#define BATCH   8
#define NPAD    8     // LDS row padding (halves)

typedef __attribute__((ext_vector_type(16))) __bf16 v16bf;
typedef __attribute__((ext_vector_type(8)))  float  v8f;

// ---- derived per-mode decay lambda = exp(dt*A) ----------------------
__device__ __forceinline__ void mode_lambda(const float* A, const float* logdt,
                                            int n, float& dt, float& lr, float& li) {
  dt = __expf(logdt[n]);
  float sp  = log1pf(__expf(A[2 * n]));   // softplus(log_A_real)
  float ar  = -dt * sp;
  float ai  =  dt * A[2 * n + 1];
  float rad = __expf(ar);
  lr = rad * __cosf(ai);
  li = rad * __sinf(ai);
}

// ---- f32 -> bf16 pre-conversion of the C projection matrix ----------
__global__ void cvt_bf16(const float* __restrict__ src, __bf16* __restrict__ dst, int n) {
  int i = blockIdx.x * blockDim.x + threadIdx.x;
  if (i < n) dst[i] = (__bf16)src[i];
}

// ---- phase 1: per-chunk end state with zero init --------------------
__global__ void ssm_carry(const float* __restrict__ x, const float* __restrict__ A,
                          const float* __restrict__ Bm, const float* __restrict__ logdt,
                          float2* __restrict__ e, int cin, int T, int nChunks) {
  int n = threadIdx.x;          // mode
  int k = blockIdx.x;           // chunk
  int b = blockIdx.y;           // batch
  float dt, lr, li;
  mode_lambda(A, logdt, n, dt, lr, li);
  const float* xb = x + (size_t)b * cin * T;
  const float* Bn = Bm + n * cin;
  int t0 = k * CHUNK;
  float hr = 0.f, hi = 0.f;
  for (int t = 0; t < CHUNK; ++t) {
    if ((t & 15) == 0)
      __builtin_prefetch(&xb[t0 + t + CHUNK], 0, 1);   // global_prefetch_b8
    float u = 0.f;
    for (int c = 0; c < cin; ++c)
      u += Bn[c] * xb[c * T + t0 + t];                  // lane-uniform x load
    u *= dt;
    float nr = lr * hr - li * hi + u;
    hi = lr * hi + li * hr;
    hr = nr;
  }
  e[((size_t)b * NSTATE + n) * nChunks + k] = make_float2(hr, hi);
}

// ---- phase 2: propagate carries: init_k = lam^64 * init_{k-1} + e_{k-1}
__global__ void ssm_prefix(const float* __restrict__ A, const float* __restrict__ logdt,
                           const float2* __restrict__ e, float2* __restrict__ hinit,
                           int nChunks) {
  int gid = blockIdx.x * blockDim.x + threadIdx.x;  // b*256+n
  if (gid >= BATCH * NSTATE) return;
  int n = gid & (NSTATE - 1);
  float dt, lr, li;
  mode_lambda(A, logdt, n, dt, lr, li);
  float pr = lr, pi = li;                 // lambda^64 via 6 squarings
  for (int s = 0; s < 6; ++s) { float t = pr * pr - pi * pi; pi = 2.f * pr * pi; pr = t; }
  const float2* eb = e + (size_t)gid * nChunks;
  float2*       hb = hinit + (size_t)gid * nChunks;
  float hr = 0.f, hi = 0.f;
  hb[0] = make_float2(0.f, 0.f);
  for (int k = 1; k < nChunks; ++k) {
    float2 ek = eb[k - 1];
    float nr = pr * hr - pi * hi + ek.x;
    hi = pr * hi + pi * hr + ek.y;
    hr = nr;
    hb[k] = make_float2(hr, hi);
  }
}

// ---- phase 3: re-scan chunk, stage Re(h) (bf16, transposed) in LDS,
//      WMMA-contract with pre-converted bf16 C --------------------------
__global__ void ssm_apply(const float* __restrict__ x, const float* __restrict__ A,
                          const float* __restrict__ Bm, const float* __restrict__ logdt,
                          const float* __restrict__ Cm, const __bf16* __restrict__ Cbf,
                          const float2* __restrict__ hinit,
                          float* __restrict__ y, int cin, int cout, int T, int nChunks) {
  __shared__ __bf16 hReT[CHUNK][NSTATE + NPAD];   // ~33 KB
  int n = threadIdx.x;
  int k = blockIdx.x;
  int b = blockIdx.y;
  float dt, lr, li;
  mode_lambda(A, logdt, n, dt, lr, li);
  const float* xb = x + (size_t)b * cin * T;
  const float* Bn = Bm + n * cin;
  int t0 = k * CHUNK;
  float2 h0 = hinit[((size_t)b * NSTATE + n) * nChunks + k];
  float hr = h0.x, hi = h0.y;
  for (int t = 0; t < CHUNK; ++t) {
    float u = 0.f;
    for (int c = 0; c < cin; ++c)
      u += Bn[c] * xb[c * T + t0 + t];
    u *= dt;
    float nr = lr * hr - li * hi + u;
    hi = lr * hi + li * hr;
    hr = nr;
    hReT[t][n] = (__bf16)hr;
  }
  __syncthreads();

  if (cout >= 16) {
    // y_tile(16d x 16t) = C(16 x 256) @ ReH(256 x 16) via 8 K-steps of 32.
    int wave = threadIdx.x >> 5;
    int lane = threadIdx.x & 31;
    int lo   = lane & 15;
    int hb2  = lane >> 4;
    const int colTiles = CHUNK / 16;                 // 4
    int nTiles = (cout >> 4) * colTiles;
    float* yb = y + (size_t)b * cout * T;            // single 64-bit base
    for (int tile = wave; tile < nTiles; tile += 8) {  // wave-uniform loop
      int dBase = (tile / colTiles) << 4;
      int tBase = (tile % colTiles) << 4;
      v8f acc = {0.f, 0.f, 0.f, 0.f, 0.f, 0.f, 0.f, 0.f};
      int M    = dBase + lo;       // output row for A-frag
      int tcol = tBase + lo;       // output col for B/D-frag
#pragma unroll
      for (int kb = 0; kb < NSTATE; kb += 32) {
        v16bf afrag, bfrag;
        // A 16x32 lane layout: two contiguous 8-half (16 B) runs.
        const __bf16* arow = Cbf + M * NSTATE + kb + (hb2 << 3);
#pragma unroll
        for (int i = 0; i < 8; ++i) { afrag[i] = arow[i]; afrag[i + 8] = arow[16 + i]; }
        // B 32x16 lane layout: two contiguous 8-half (16 B) runs in LDS.
        const __bf16* brow = &hReT[tcol][kb + (hb2 << 3)];
#pragma unroll
        for (int i = 0; i < 8; ++i) { bfrag[i] = brow[i]; bfrag[i + 8] = brow[16 + i]; }
        acc = __builtin_amdgcn_wmma_f32_16x16x32_bf16(
            false, afrag, false, bfrag, (short)0, acc, false, false);
      }
#pragma unroll
      for (int v = 0; v < 8; ++v) {                  // C/D f32 lane layout
        int Md = dBase + v + (hb2 << 3);
        yb[Md * T + t0 + tcol] = acc[v];             // 32-bit offset math
      }
    }
  } else {
    // cout == 1: plain 256-wide dot per timestep.
    if (threadIdx.x < CHUNK) {
      int t = threadIdx.x;
      float s = 0.f;
      for (int nn = 0; nn < NSTATE; ++nn) s += Cm[nn] * (float)hReT[t][nn];
      y[(size_t)b * T + t0 + t] = s;
    }
  }
}

// ---- depthwise conv1d, k=3, pad=1 (cross-correlation, jax lax.conv) --
__global__ void dwconv3(const float* __restrict__ x, const float* __restrict__ w,
                        const float* __restrict__ bias, float* __restrict__ y,
                        int C, int T) {
  int idx = blockIdx.x * blockDim.x + threadIdx.x;
  int tot = BATCH * C * T;
  if (idx >= tot) return;
  int t = idx % T;
  int c = (idx / T) % C;
  size_t base = (size_t)(idx / T) * T;
  float xm = (t > 0)     ? x[base + t - 1] : 0.f;
  float x0 = x[base + t];
  float xp = (t < T - 1) ? x[base + t + 1] : 0.f;
  y[base + t] = w[c * 3] * xm + w[c * 3 + 1] * x0 + w[c * 3 + 2] * xp + bias[c];
}

// ---- LayerNorm over channels (optional) + SiLU, in place ------------
__global__ void ln_silu(float* __restrict__ x, const float* __restrict__ w,
                        const float* __restrict__ bsh, int hasln, int C, int T) {
  int idx = blockIdx.x * blockDim.x + threadIdx.x;
  if (idx >= BATCH * T) return;
  int b = idx / T, t = idx % T;
  float* col = x + (size_t)b * C * T + t;
  if (hasln) {
    float mu = 0.f, m2 = 0.f;
    for (int c = 0; c < C; ++c) { float v = col[c * T]; mu += v; m2 += v * v; }
    mu /= (float)C;
    float var = m2 / (float)C - mu * mu;
    float inv = rsqrtf(var + 1e-5f);
    for (int c = 0; c < C; ++c) {
      float v = (col[c * T] - mu) * inv * w[c] + bsh[c];
      col[c * T] = v / (1.f + __expf(-v));
    }
  } else {
    for (int c = 0; c < C; ++c) {
      float v = col[c * T];
      col[c * T] = v / (1.f + __expf(-v));
    }
  }
}

// ---- y[b,d,t] = sum_{c,r} x[b,c,t*R+r] * W[c,d,r] -------------------
__global__ void mix_down(const float* __restrict__ x, const float* __restrict__ W,
                         float* __restrict__ y, int Cin, int Cout, int Tin, int R) {
  int Tout = Tin / R;
  int idx = blockIdx.x * blockDim.x + threadIdx.x;
  int tot = BATCH * Cout * Tout;
  if (idx >= tot) return;
  int t = idx % Tout;
  int d = (idx / Tout) % Cout;
  int b = idx / (Tout * Cout);
  float acc = 0.f;
  for (int c = 0; c < Cin; ++c) {
    const float* xr = x + ((size_t)b * Cin + c) * Tin + t * R;
    const float* wr = W + (c * Cout + d) * R;
    for (int r = 0; r < R; ++r) acc += xr[r] * wr[r];
  }
  y[((size_t)b * Cout + d) * Tout + t] = acc;
}

// ---- y[b,d,t*R+rr] = skip + sum_c x[b,c,t] * W[c,d,rr] --------------
__global__ void mix_up_skip(const float* __restrict__ x, const float* __restrict__ W,
                            const float* __restrict__ skip, float* __restrict__ y,
                            int Cin, int Cout, int Tin, int R) {
  int Ttot = Tin * R;
  int idx = blockIdx.x * blockDim.x + threadIdx.x;
  int tot = BATCH * Cout * Ttot;
  if (idx >= tot) return;
  int tt = idx % Ttot;
  int d  = (idx / Ttot) % Cout;
  int b  = idx / (Ttot * Cout);
  int t  = tt / R, rr = tt % R;
  float acc = skip[((size_t)b * Cout + d) * Ttot + tt];
  for (int c = 0; c < Cin; ++c)
    acc += x[((size_t)b * Cin + c) * Tin + t] * W[(c * Cout + d) * R + rr];
  y[((size_t)b * Cout + d) * Ttot + tt] = acc;
}

// =====================================================================
// Host orchestration
// =====================================================================
struct SsmP   { const float *A, *B, *logdt, *C; };
struct BlockP { bool conv, ln; const float *cw, *cb, *lnw, *lnb; SsmP s; };

extern "C" void kernel_launch(void* const* d_in, const int* in_sizes, int n_in,
                              void* d_out, int out_size, void* d_ws, size_t ws_size,
                              hipStream_t stream) {
  (void)in_sizes; (void)n_in; (void)out_size; (void)ws_size;
  const int CH[7] = {1, 16, 32, 64, 96, 128, 256};
  const int RF[6] = {4, 4, 2, 2, 2, 2};
  int Tl[7]; Tl[0] = 65536;
  for (int i = 0; i < 6; ++i) Tl[i + 1] = Tl[i] / RF[i];

  int cur = 0;
  auto nxt = [&]() { return (const float*)d_in[cur++]; };
  auto parseSsm = [&]() { SsmP s; s.A = nxt(); s.B = nxt(); s.logdt = nxt(); s.C = nxt(); return s; };
  auto parseBlock = [&](int c, bool act) {
    BlockP b{};
    if (c > 1) { b.conv = true; b.cw = nxt(); b.cb = nxt(); }
    b.s = parseSsm();
    if (act && c > 1) { b.ln = true; b.lnw = nxt(); b.lnb = nxt(); }
    return b;
  };

  const float* X = nxt();
  BlockP dblk[6]; const float* dmix[6];
  for (int i = 0; i < 6; ++i) { dblk[i] = parseBlock(CH[i], true); dmix[i] = nxt(); }
  const float* umix[6]; BlockP ublk[6];
  for (int i = 0; i < 6; ++i) { umix[i] = nxt(); ublk[i] = parseBlock(CH[i], true); }
  BlockP hblk[2] = {parseBlock(256, true), parseBlock(256, true)};
  BlockP lblk[2] = {parseBlock(1, true),  parseBlock(1, false)};

  // ---- workspace carve-up (floats) ----
  float* wsf = (float*)d_ws;
  const size_t ACT = 2097152;              // max activation = (8,16,16384)
  float* actA = wsf;
  float* actB = wsf + ACT;
  float* actC = wsf + 2 * ACT;
  size_t skipSz[6] = {524288, 2097152, 1048576, 1048576, 786432, 524288};
  float* skipP[6];
  {
    float* p = wsf + 3 * ACT;
    for (int i = 0; i < 6; ++i) { skipP[i] = p; p += skipSz[i]; }
  }
  size_t skipTot = 0; for (int i = 0; i < 6; ++i) skipTot += skipSz[i];
  float2* eBuf = (float2*)(wsf + 3 * ACT + skipTot);
  float2* hBuf = eBuf + (size_t)BATCH * NSTATE * (65536 / CHUNK);
  __bf16* Cbf  = (__bf16*)(hBuf + (size_t)BATCH * NSTATE * (65536 / CHUNK));

  auto runSsm = [&](const SsmP& s, const float* xin, float* yout, int c, int T) {
    int nCh = T / CHUNK;
    dim3 g(nCh, BATCH);
    int cElems = c * NSTATE;
    cvt_bf16  <<<(cElems + 255) / 256, 256, 0, stream>>>(s.C, Cbf, cElems);
    ssm_carry <<<g, NSTATE, 0, stream>>>(xin, s.A, s.B, s.logdt, eBuf, c, T, nCh);
    ssm_prefix<<<(BATCH * NSTATE + 255) / 256, 256, 0, stream>>>(s.A, s.logdt, eBuf, hBuf, nCh);
    ssm_apply <<<g, NSTATE, 0, stream>>>(xin, s.A, s.B, s.logdt, s.C, Cbf, hBuf, yout,
                                         c, c, T, nCh);
  };
  auto runBlock = [&](const BlockP& blk, const float* xin, float* yout, float* tmp,
                      int c, int T, bool act) {
    const float* curp = xin;
    if (blk.conv) {
      int tot = BATCH * c * T;
      dwconv3<<<(tot + 255) / 256, 256, 0, stream>>>(xin, blk.cw, blk.cb, tmp, c, T);
      curp = tmp;
    }
    runSsm(blk.s, curp, yout, c, T);
    if (act) {
      int tot = BATCH * T;
      ln_silu<<<(tot + 255) / 256, 256, 0, stream>>>(yout, blk.lnw, blk.lnb,
                                                     blk.ln ? 1 : 0, c, T);
    }
  };

  // ---- encoder ----
  const float* h = X;
  for (int i = 0; i < 6; ++i) {
    hipMemcpyAsync(skipP[i], h, (size_t)BATCH * CH[i] * Tl[i] * sizeof(float),
                   hipMemcpyDeviceToDevice, stream);
    runBlock(dblk[i], h, actB, actC, CH[i], Tl[i], true);
    int tot = BATCH * CH[i + 1] * Tl[i + 1];
    mix_down<<<(tot + 255) / 256, 256, 0, stream>>>(actB, dmix[i], actA,
                                                    CH[i], CH[i + 1], Tl[i], RF[i]);
    h = actA;
  }
  // ---- bottleneck ----
  runBlock(hblk[0], h,    actB, actC, 256, Tl[6], true);
  runBlock(hblk[1], actB, actA, actC, 256, Tl[6], true);
  h = actA;
  // ---- decoder ----
  for (int i = 5; i >= 0; --i) {
    int tot = BATCH * CH[i] * Tl[i];
    mix_up_skip<<<(tot + 255) / 256, 256, 0, stream>>>(h, umix[i], skipP[i], actB,
                                                       CH[i + 1], CH[i], Tl[i + 1], RF[i]);
    runBlock(ublk[i], actB, actA, actC, CH[i], Tl[i], true);
    h = actA;
  }
  // ---- head ----
  runBlock(lblk[0], h,    actB,           actC, 1, Tl[0], true);
  runBlock(lblk[1], actB, (float*)d_out,  actC, 1, Tl[0], false);
}
